// LIFReadout_58841051955590
// MI455X (gfx1250) — matrix operand order repaired and checked by reading, not compiled, not run on hardware
//
#include <hip/hip_runtime.h>
#include <stdint.h>

typedef uint32_t u32;
typedef uint64_t u64;
typedef __attribute__((ext_vector_type(4)))  unsigned int uint32x4;
typedef __attribute__((ext_vector_type(8)))  int          int32x8;
typedef __attribute__((ext_vector_type(4)))  int          int32x4;
typedef __attribute__((ext_vector_type(16))) __bf16       v16bf;
typedef __attribute__((ext_vector_type(8)))  __bf16       v8bf;
typedef __attribute__((ext_vector_type(8)))  float        v8f;

#define T_STEPS 100
#define BB      128
#define CC      2048
#define OO      1000
#define DECAY   0.9f
#define THRESH  1.0f

#define KCHUNK  128
#define NCH     (CC / KCHUNK)          // 16 K-chunks per timestep
#define BTILE   32                     // B rows per block
#define OTILE   32                     // O cols per block
#define WAVES   4                      // 2x2 waves over (rows, cols)

// ---- shared LDS geometry -----------------------------------------------------
// fast path:   [0, 2*FABUF)  A ping-pong, each = AH(8KB) + AL(8KB) bf16 32x128
// fallback:    [0, 2*ABYTES) A ping-pong, each = 32x128 f32 (16KB)
// both:        persistent W-hi / W-lo bf16, column-major [OTILE][CC]
#define ABYTES   (BTILE * KCHUNK * 4)  // 16384 (f32 tile / also fast-path buf)
#define AHALF    (BTILE * KCHUNK * 2)  // 8192  (bf16 plane)
#define WHI_OFF  (2 * ABYTES)          // 32768
#define WBYTES   (OTILE * CC * 2)      // 131072
#define WLO_OFF  (WHI_OFF + WBYTES)    // 163840
#define LDSBYTES (WLO_OFF + WBYTES)    // 294912 (288 KB <= 320 KB/WGP)

#define DIM_BIG  (1u << 20)            // "never OOB" tensor dimension

#define XELEMS   ((u64)T_STEPS * BB * CC)     // 26,214,400
#define XPLANE   (XELEMS * 2)                 // bytes per bf16 plane

__device__ __forceinline__ float truncbf(float v) {
    return __builtin_bit_cast(float, __builtin_bit_cast(u32, v) & 0xFFFF0000u);
}

// Issue a 2-D TDM tile load (D# bitfields per CDNA5 ISA 8.3/8.4).
// dsz: data_size code (1 = 2B, 2 = 4B); dims/strides/tiles in element units.
__device__ __forceinline__ void tdm_load_2d(u32 lds_off, u64 gaddr,
                                            u32 dim0, u32 dim1,
                                            u32 tile0, u32 tile1, u64 stride0,
                                            u32 dsz) {
    uint32x4 g0;
    g0[0] = 1u;                                   // count=1 (valid), user mode
    g0[1] = lds_off;                              // lds_addr [63:32]
    g0[2] = (u32)(gaddr & 0xFFFFFFFFull);         // global_addr [95:64]
    g0[3] = (u32)((gaddr >> 32) & 0x01FFFFFFull)  // global_addr [120:96]
          | (2u << 30);                           // type=2 ("image") [127:126]
    int32x8 g1;
    g1[0] = (int)(dsz << 16);                     // workgroup_mask=0, data_size
    g1[1] = (int)((dim0 & 0xFFFFu) << 16);        // tensor_dim0[15:0] @ [63:48]
    g1[2] = (int)(((dim0 >> 16) & 0xFFFFu) | ((dim1 & 0xFFFFu) << 16));
    g1[3] = (int)(((dim1 >> 16) & 0xFFFFu) | (tile0 << 16));   // tile_dim0
    g1[4] = (int)(tile1 & 0xFFFFu);               // tile_dim1, tile_dim2=0
    g1[5] = (int)(stride0 & 0xFFFFFFFFull);       // tensor_dim0_stride[31:0]
    g1[6] = (int)((stride0 >> 32) & 0xFFFFu);     // stride0[47:32], stride1=0
    g1[7] = 0;
    int32x4 z4 = {0, 0, 0, 0};
    int32x8 z8 = {0, 0, 0, 0, 0, 0, 0, 0};
    __builtin_amdgcn_tensor_load_to_lds(g0, g1, z4, z4, z8, 0);
}

// Convert this block's W slab -> bf16 hi/lo, persistent LDS, column-major [col][K].
__device__ __forceinline__ void convert_w_to_lds(float* lds, const float* W,
                                                 int tid, int o0) {
    u32* wh32 = (u32*)((char*)lds + WHI_OFF);
    u32* wl32 = (u32*)((char*)lds + WLO_OFF);
    const int c    = tid & 31;
    const int kq   = tid >> 5;
    const int gcol = o0 + c;
    for (int k = kq * (CC / 4); k < (kq + 1) * (CC / 4); k += 2) {
        float v0 = 0.0f, v1 = 0.0f;
        if (gcol < OO) {
            v0 = W[(u64)k * OO + gcol];
            v1 = W[(u64)(k + 1) * OO + gcol];
        }
        const float h0 = truncbf(v0), h1 = truncbf(v1);
        const u32 hp = (__builtin_bit_cast(u32, h0) >> 16)
                     | (__builtin_bit_cast(u32, h1) & 0xFFFF0000u);
        const __bf16 l0 = (__bf16)(v0 - h0), l1 = (__bf16)(v1 - h1);
        const u32 lp = (u32)__builtin_bit_cast(unsigned short, l0)
                     | ((u32)__builtin_bit_cast(unsigned short, l1) << 16);
        const int idx = c * (CC / 2) + (k >> 1);
        wh32[idx] = hp;
        wl32[idx] = lp;
    }
}

// ---- phase 1 (fast path): split x into bf16 hi/lo planes ---------------------
__global__ __launch_bounds__(256)
void split_x_kernel(const float* __restrict__ x, u32* __restrict__ xhi,
                    u32* __restrict__ xlo) {
    const u64 i = (u64)blockIdx.x * blockDim.x + threadIdx.x;   // pair index
    if (i >= XELEMS / 2) return;
    const float2 v = ((const float2*)x)[i];
    const float h0 = truncbf(v.x), h1 = truncbf(v.y);
    const u32 hp = (__builtin_bit_cast(u32, h0) >> 16)
                 | (__builtin_bit_cast(u32, h1) & 0xFFFF0000u);
    const __bf16 l0 = (__bf16)(v.x - h0), l1 = (__bf16)(v.y - h1);
    const u32 lp = (u32)__builtin_bit_cast(unsigned short, l0)
                 | ((u32)__builtin_bit_cast(unsigned short, l1) << 16);
    xhi[i] = hp;
    xlo[i] = lp;
}

// ---- fast path main kernel: pre-split bf16 x, zero-VALU operand feed ---------
__global__ __launch_bounds__(WAVES * 32)
void lif_fast_kernel(const __bf16* __restrict__ xhi, const __bf16* __restrict__ xlo,
                     const float* __restrict__ W, const float* __restrict__ bias,
                     float* __restrict__ out) {
    extern __shared__ float lds[];

    const int tid   = threadIdx.x;
    const int wave  = tid >> 5;
    const int lane  = tid & 31;
    const int n     = lane & 15;
    const int halfw = lane >> 4;
    const int rh    = wave >> 1;
    const int ch    = wave & 1;

    const int b0       = blockIdx.y * BTILE;
    const int o0       = blockIdx.x * OTILE;
    const int colLocal = ch * 16 + n;
    const int col      = o0 + colLocal;

    const u64 xha = (u64)(uintptr_t)xhi;
    const u64 xla = (u64)(uintptr_t)xlo;

    float bval = 0.0f;
    if (col < OO) bval = bias[col] + (THRESH / (2.0f * (float)T_STEPS));

    v8f mem = {}, spk = {}, acc = {};
    const int NC = T_STEPS * NCH;

    // wave0 DMAs the hi plane, wave1 the lo plane (bf16, 32x128 tile).
    auto issue = [&](int c) {
        const int t  = c / NCH;
        const int k0 = (c % NCH) * KCHUNK;
        const u64 eoff = ((u64)(t * BB + b0)) * (u64)CC + (u64)k0;   // elements
        const u32 lb   = (u32)((c & 1) * ABYTES);
        if (wave == 0)
            tdm_load_2d(lb,         xha + eoff * 2, DIM_BIG, DIM_BIG,
                        KCHUNK, BTILE, CC, 1);
        else if (wave == 1)
            tdm_load_2d(lb + AHALF, xla + eoff * 2, DIM_BIG, DIM_BIG,
                        KCHUNK, BTILE, CC, 1);
    };

    issue(0);                                // overlap with W conversion
    convert_w_to_lds(lds, W, tid, o0);
    __syncthreads();

    const __bf16* wh = (const __bf16*)((const char*)lds + WHI_OFF);
    const __bf16* wl = (const __bf16*)((const char*)lds + WLO_OFF);

    for (int c = 0; c < NC; ++c) {
        if (c + 1 < NC) {
            issue(c + 1);
            __builtin_amdgcn_s_wait_tensorcnt(1);
        } else {
            __builtin_amdgcn_s_wait_tensorcnt(0);
        }
        __syncthreads();

        const __bf16* Ah = (const __bf16*)((const char*)lds + (c & 1) * ABYTES);
        const __bf16* Al = (const __bf16*)((const char*)Ah + AHALF);
        const int k0   = (c % NCH) * KCHUNK;
        const int arow = rh * 16 + n;

        #pragma unroll
        for (int ks = 0; ks < KCHUNK / 32; ++ks) {
            const int kb = ks * 32;

            // A layout: halves 0..15 = K kb+16*halfw+0..15 (contiguous, matches B).
            // Two contiguous 16B runs -> 2x ds_load_b128 each, shuffled together.
            const int abase = arow * KCHUNK + kb + 16 * halfw;
            const v8bf ah0 = *(const v8bf*)(Ah + abase);
            const v8bf ah1 = *(const v8bf*)(Ah + abase + 8);
            const v8bf al0 = *(const v8bf*)(Al + abase);
            const v8bf al1 = *(const v8bf*)(Al + abase + 8);
            const v16bf ahi = __builtin_shufflevector(ah0, ah1,
                0, 1, 2, 3, 4, 5, 6, 7, 8, 9, 10, 11, 12, 13, 14, 15);
            const v16bf alo = __builtin_shufflevector(al0, al1,
                0, 1, 2, 3, 4, 5, 6, 7, 8, 9, 10, 11, 12, 13, 14, 15);

            // B layout: halves 0..15 = K kb+16*halfw+0..15, contiguous 32B.
            const u64 bidx = (u64)colLocal * CC + (u64)(k0 + kb + 16 * halfw);
            const v16bf bhi = *(const v16bf*)(wh + bidx);
            const v16bf blo = *(const v16bf*)(wl + bidx);

            acc = __builtin_amdgcn_wmma_f32_16x16x32_bf16(false, ahi, false, bhi,
                                                          (short)0, acc, false, false);
            acc = __builtin_amdgcn_wmma_f32_16x16x32_bf16(false, alo, false, bhi,
                                                          (short)0, acc, false, false);
            acc = __builtin_amdgcn_wmma_f32_16x16x32_bf16(false, ahi, false, blo,
                                                          (short)0, acc, false, false);
        }

        if ((c & (NCH - 1)) == (NCH - 1)) {
            #pragma unroll
            for (int r = 0; r < 8; ++r) {
                const float cur = acc[r] + bval;
                const float m   = DECAY * mem[r] + cur;
                const float s   = (m > THRESH) ? 1.0f : 0.0f;
                mem[r] = m - s * THRESH;
                spk[r] += s;
                acc[r] = 0.0f;
            }
        }
        __syncthreads();
    }

    if (col < OO) {
        const float inv = 1.0f / (float)T_STEPS;
        #pragma unroll
        for (int r = 0; r < 8; ++r) {
            const int row = b0 + rh * 16 + r + 8 * halfw;
            out[row * OO + col] = spk[r] * inv;
        }
    }
}

// ---- fallback: in-loop x split (no workspace needed) -------------------------
__global__ __launch_bounds__(WAVES * 32)
void lif_fused_kernel(const float* __restrict__ x, const float* __restrict__ W,
                      const float* __restrict__ bias, float* __restrict__ out) {
    extern __shared__ float lds[];

    const int tid   = threadIdx.x;
    const int wave  = tid >> 5;
    const int lane  = tid & 31;
    const int n     = lane & 15;
    const int halfw = lane >> 4;
    const int rh    = wave >> 1;
    const int ch    = wave & 1;

    const int b0       = blockIdx.y * BTILE;
    const int o0       = blockIdx.x * OTILE;
    const int colLocal = ch * 16 + n;
    const int col      = o0 + colLocal;

    const u64 xaddr = (u64)(uintptr_t)x;

    float bval = 0.0f;
    if (col < OO) bval = bias[col] + (THRESH / (2.0f * (float)T_STEPS));

    v8f mem = {}, spk = {}, acc = {};
    const int NC = T_STEPS * NCH;

    auto issue = [&](int c) {
        if (wave == 0) {
            const int t  = c / NCH;
            const int k0 = (c % NCH) * KCHUNK;
            const u64 ga = xaddr + ((((u64)(t * BB + b0)) * (u64)CC + (u64)k0) << 2);
            tdm_load_2d((u32)((c & 1) * ABYTES), ga, DIM_BIG, DIM_BIG,
                        KCHUNK, BTILE, CC, 2);
        }
    };

    issue(0);
    convert_w_to_lds(lds, W, tid, o0);
    __syncthreads();

    const __bf16* wh = (const __bf16*)((const char*)lds + WHI_OFF);
    const __bf16* wl = (const __bf16*)((const char*)lds + WLO_OFF);

    for (int c = 0; c < NC; ++c) {
        if (c + 1 < NC) {
            issue(c + 1);
            __builtin_amdgcn_s_wait_tensorcnt(1);
        } else {
            __builtin_amdgcn_s_wait_tensorcnt(0);
        }
        __syncthreads();

        const float* Abuf = (const float*)((const char*)lds + (c & 1) * ABYTES);
        const int k0   = (c % NCH) * KCHUNK;
        const int arow = rh * 16 + n;

        #pragma unroll
        for (int ks = 0; ks < KCHUNK / 32; ++ks) {
            const int kb = ks * 32;
            v16bf ahi, alo;
            #pragma unroll
            for (int j = 0; j < 8; ++j) {
                const int k = kb + 16 * halfw + 2 * j;
                const float2 av = *(const float2*)(Abuf + arow * KCHUNK + k);
                const float h0 = truncbf(av.x), h1 = truncbf(av.y);
                ahi[2 * j]     = (__bf16)h0;
                ahi[2 * j + 1] = (__bf16)h1;
                alo[2 * j]     = (__bf16)(av.x - h0);
                alo[2 * j + 1] = (__bf16)(av.y - h1);
            }
            const u64 bidx = (u64)colLocal * CC + (u64)(k0 + kb + 16 * halfw);
            const v16bf bhi = *(const v16bf*)(wh + bidx);
            const v16bf blo = *(const v16bf*)(wl + bidx);

            acc = __builtin_amdgcn_wmma_f32_16x16x32_bf16(false, ahi, false, bhi,
                                                          (short)0, acc, false, false);
            acc = __builtin_amdgcn_wmma_f32_16x16x32_bf16(false, alo, false, bhi,
                                                          (short)0, acc, false, false);
            acc = __builtin_amdgcn_wmma_f32_16x16x32_bf16(false, ahi, false, blo,
                                                          (short)0, acc, false, false);
        }

        if ((c & (NCH - 1)) == (NCH - 1)) {
            #pragma unroll
            for (int r = 0; r < 8; ++r) {
                const float cur = acc[r] + bval;
                const float m   = DECAY * mem[r] + cur;
                const float s   = (m > THRESH) ? 1.0f : 0.0f;
                mem[r] = m - s * THRESH;
                spk[r] += s;
                acc[r] = 0.0f;
            }
        }
        __syncthreads();
    }

    if (col < OO) {
        const float inv = 1.0f / (float)T_STEPS;
        #pragma unroll
        for (int r = 0; r < 8; ++r) {
            const int row = b0 + rh * 16 + r + 8 * halfw;
            out[row * OO + col] = spk[r] * inv;
        }
    }
}

extern "C" void kernel_launch(void* const* d_in, const int* in_sizes, int n_in,
                              void* d_out, int out_size, void* d_ws, size_t ws_size,
                              hipStream_t stream) {
    (void)in_sizes; (void)n_in; (void)out_size;
    const float* x = (const float*)d_in[0];   // [100,128,2048] f32
    const float* W = (const float*)d_in[1];   // [2048,1000]    f32
    const float* b = (const float*)d_in[2];   // [1000]         f32
    float* out = (float*)d_out;               // [128,1000]     f32

    dim3 grid((OO + OTILE - 1) / OTILE, BB / BTILE, 1);   // 32 x 4 workgroups
    dim3 block(WAVES * 32, 1, 1);                          // 4 waves (wave32)

    if (ws_size >= 2 * XPLANE && d_ws != nullptr) {
        u32* xhi = (u32*)d_ws;
        u32* xlo = (u32*)((char*)d_ws + XPLANE);
        const u64 npairs = XELEMS / 2;
        split_x_kernel<<<(unsigned)((npairs + 255) / 256), 256, 0, stream>>>(
            x, xhi, xlo);
        lif_fast_kernel<<<grid, block, LDSBYTES, stream>>>(
            (const __bf16*)xhi, (const __bf16*)xlo, W, b, out);
    } else {
        lif_fused_kernel<<<grid, block, LDSBYTES, stream>>>(x, W, b, out);
    }
}
